// QAwareBertSelfAttention_41429254537233
// MI455X (gfx1250) — matrix operand order repaired and checked by reading, not compiled, not run on hardware
//
#include <hip/hip_runtime.h>
#include <cstdint>

#define BATCH 8
#define SEQ   512
#define DM    1024
#define NH    16
#define HD    64
#define TOKENS (BATCH*SEQ)     // 4096
#define HEADS  (BATCH*NH)      // 128

typedef __attribute__((ext_vector_type(16))) __bf16 v16bf;
typedef __attribute__((ext_vector_type(8)))  float  v8f;

union FragU { v16bf v; unsigned u[8]; };

// float -> bf16, round to nearest even
__device__ __forceinline__ unsigned short f2bf(float f) {
  unsigned u = __float_as_uint(f);
  unsigned r = 0x7FFFu + ((u >> 16) & 1u);
  return (unsigned short)((u + r) >> 16);
}

// order-preserving float<->uint encoding so atomicMin/Max on u32 give float min/max
__device__ __forceinline__ unsigned f2ord(float f) {
  unsigned u = __float_as_uint(f);
  return (u & 0x80000000u) ? ~u : (u | 0x80000000u);
}
__device__ __forceinline__ float ord2f(unsigned u) {
  return __uint_as_float((u & 0x80000000u) ? (u & 0x7FFFFFFFu) : ~u);
}

__device__ __forceinline__ float wredMax(float v){ for(int o=16;o;o>>=1) v=fmaxf(v,__shfl_xor(v,o)); return v; }
__device__ __forceinline__ float wredMin(float v){ for(int o=16;o;o>>=1) v=fminf(v,__shfl_xor(v,o)); return v; }
__device__ __forceinline__ float wredSum(float v){ for(int o=16;o;o>>=1) v+=__shfl_xor(v,o); return v; }

// A fragment: 16x32 bf16 tile from row-major source (lda in elements).
// ISA layout: lanes 0-15 hold M=lane, K pairs {kb+0..7, kb+16..23}, kb=8 for lanes 16-31.
__device__ __forceinline__ v16bf load_fragA(const unsigned short* A, int lda, int lane) {
  const unsigned short* p = A + (size_t)(lane & 15) * lda + ((lane >> 4) << 3);
  FragU f;
#pragma unroll
  for (int i = 0; i < 4; ++i) {
    f.u[i]     = *(const unsigned*)(p + 2*i);
    f.u[i + 4] = *(const unsigned*)(p + 16 + 2*i);
  }
  return f.v;
}

// B fragment: 32x16 bf16 tile; source is row-major with rows = columns of B (Bsrc[n][k]).
// ISA layout: lanes 0-15 hold N=lane, K {kb+0..7, kb+16..23}, kb=0; kb=8 for lanes 16-31.
__device__ __forceinline__ v16bf load_fragB(const unsigned short* Bsrc, int ldb, int lane) {
  const unsigned short* p = Bsrc + (size_t)(lane & 15) * ldb + ((lane >> 4) << 3);
  FragU f;
#pragma unroll
  for (int i = 0; i < 4; ++i) {
    f.u[i]     = *(const unsigned*)(p + 2*i);
    f.u[i + 4] = *(const unsigned*)(p + 16 + 2*i);
  }
  return f.v;
}

// 64x64 macro-tile GEMM: 4x4 register blocking, 16 accumulators per wave.
// 8 fragment loads (8KB) feed 16 WMMAs per K-step -> 0.5KB/WMMA of L2 traffic.
// K-loop kept rolled: one iteration already holds 16 WMMAs; unrolling only
// multiplies live fragment ranges and causes spills.
__device__ __forceinline__ void gemm64x64(const unsigned short* __restrict__ A, int lda,
                                          const unsigned short* __restrict__ B, int ldb,
                                          int K, int lane, v8f c[16]) {
#pragma unroll 1
  for (int kk = 0; kk < K; kk += 32) {
    v16bf af[4], bf[4];
#pragma unroll
    for (int i = 0; i < 4; ++i) af[i] = load_fragA(A + (size_t)i * 16 * lda + kk, lda, lane);
#pragma unroll
    for (int j = 0; j < 4; ++j) bf[j] = load_fragB(B + (size_t)j * 16 * ldb + kk, ldb, lane);
#pragma unroll
    for (int i = 0; i < 4; ++i)
#pragma unroll
      for (int j = 0; j < 4; ++j)
        c[i*4+j] = __builtin_amdgcn_wmma_f32_16x16x32_bf16(false, af[i], false, bf[j],
                                                           (short)0, c[i*4+j], false, false);
  }
}

__global__ void k_convert_bf16(const float* __restrict__ in, unsigned short* __restrict__ out, int n) {
  int i = blockIdx.x * blockDim.x + threadIdx.x;
  if (i < n) out[i] = f2bf(in[i]);
}

__global__ void k_init_minmax(unsigned* mm) {
  int i = threadIdx.x;
  if (i < 3) { mm[2*i] = 0xFFFFFFFFu; mm[2*i+1] = 0u; }   // min-slot = +inf code, max-slot = -inf code
}

// Y = X @ W^T + b for Wq/Wk/Wv; q,k written [B,H,S,D] bf16, v written transposed [B,H,D,S] bf16
// One 64x64 macro-tile per wave. launch_bounds(...,1): allow full VGPR budget, no spills.
__global__ void __launch_bounds__(256, 1)
k_qkv_gemm(const unsigned short* __restrict__ X,
           const unsigned short* __restrict__ Wq,
           const unsigned short* __restrict__ Wk,
           const unsigned short* __restrict__ Wv,
           const float* __restrict__ bq,
           const float* __restrict__ bk,
           const float* __restrict__ bv,
           unsigned short* __restrict__ q,
           unsigned short* __restrict__ k,
           unsigned short* __restrict__ vT) {
  int wave = threadIdx.x >> 5, lane = threadIdx.x & 31;
  int gid = blockIdx.x * 8 + wave;
  const int NT  = DM / 64;                 // 16 macro-tiles across features
  const int TPM = (TOKENS / 64) * NT;      // 1024 macro-tiles per matrix
  int m  = gid / TPM;
  int t  = gid % TPM;
  int tm = t / NT, tn = t % NT;
  const unsigned short* W   = (m == 0) ? Wq : (m == 1) ? Wk : Wv;
  const float* bias         = (m == 0) ? bq : (m == 1) ? bk : bv;
  v8f c[16] = {};
  gemm64x64(X + (size_t)tm * 64 * DM, DM, W + (size_t)tn * 64 * DM, DM, DM, lane, c);

  int hi8 = (lane >> 4) << 3;
#pragma unroll
  for (int j = 0; j < 4; ++j) {
    int o = tn * 64 + j * 16 + (lane & 15);
    int h = o >> 6, d = o & 63;
    float bia = bias[o];
#pragma unroll
    for (int i = 0; i < 4; ++i) {
#pragma unroll
      for (int v2 = 0; v2 < 8; ++v2) {
        int n  = tm * 64 + i * 16 + hi8 + v2;
        int b_ = n >> 9, s = n & 511;
        unsigned short val = f2bf(c[i*4+j][v2] + bia);
        if (m == 0)      q [((size_t)(b_*NH + h)*SEQ + s)*HD + d] = val;
        else if (m == 1) k [((size_t)(b_*NH + h)*SEQ + s)*HD + d] = val;
        else             vT[((size_t)(b_*NH + h)*HD  + d)*SEQ + s] = val;
      }
    }
  }
}

// scores[head][sq][sk] = q . k  (per head), 64x64 macro-tile per wave, fused global min/max
__global__ void __launch_bounds__(256, 1)
k_scores_gemm(const unsigned short* __restrict__ q,
              const unsigned short* __restrict__ kmat,
              float* __restrict__ scores,
              unsigned* __restrict__ mm) {
  int wave = threadIdx.x >> 5, lane = threadIdx.x & 31;
  int gid = blockIdx.x * 8 + wave;
  const int TPH = (SEQ/64) * (SEQ/64);     // 64 macro-tiles per head
  int head = gid / TPH;
  int t = gid % TPH;
  int tm = t >> 3, tn = t & 7;
  v8f c[16] = {};
  gemm64x64(q    + (size_t)head * SEQ * HD + (size_t)tm * 64 * HD, HD,
            kmat + (size_t)head * SEQ * HD + (size_t)tn * 64 * HD, HD, HD, lane, c);

  float* out = scores + (size_t)head * SEQ * SEQ;
  int hi8 = (lane >> 4) << 3;
  float mn = c[0][0], mx = c[0][0];
#pragma unroll
  for (int i = 0; i < 4; ++i)
#pragma unroll
    for (int j = 0; j < 4; ++j)
#pragma unroll
      for (int v2 = 0; v2 < 8; ++v2) {
        float val = c[i*4+j][v2];
        out[(size_t)(tm*64 + i*16 + hi8 + v2) * SEQ + tn*64 + j*16 + (lane & 15)] = val;
        mn = fminf(mn, val); mx = fmaxf(mx, val);
      }
  mn = wredMin(mn); mx = wredMax(mx);
  if (lane == 0) { atomicMin(&mm[0], f2ord(mn)); atomicMax(&mm[1], f2ord(mx)); }
}

// compute (scale, zero_point) from encoded min/max, per reference fake_quant
__global__ void k_finalize(const unsigned* __restrict__ mm, float2* __restrict__ prm) {
  if (threadIdx.x == 0) {
    float mn = fminf(ord2f(mm[0]), 0.0f);
    float mx = fmaxf(ord2f(mm[1]), 0.0f);
    float scale = fmaxf((mx - mn) * (1.0f/255.0f), 1e-8f);
    float zp = rintf(fminf(fmaxf(-mn/scale, 0.0f), 255.0f));
    *prm = make_float2(scale, zp);
  }
}

// in-place: fake_quant(scores) / 8 + mask -> softmax -> probs; fused probs min/max
__global__ void k_softmax(float* __restrict__ scores,
                          const float* __restrict__ mask,
                          const float2* __restrict__ sprm,
                          unsigned* __restrict__ mm) {
  __shared__ float sA[8], sB[8], sC[8], sD[8];
  int row = blockIdx.x;                    // ((b*NH + h)*SEQ + sq)
  int tid = threadIdx.x, wave = tid >> 5, lane = tid & 31;
  int b = row >> 13;                       // row / (NH*SEQ)
  float scale = sprm->x, zp = sprm->y;
  float* rp = scores + (size_t)row * SEQ;
  const float* mp = mask + b * SEQ;
  float x0 = rp[tid], x1 = rp[tid + 256];
  float q0 = fminf(fmaxf(rintf(x0/scale + zp), 0.f), 255.f);
  float q1 = fminf(fmaxf(rintf(x1/scale + zp), 0.f), 255.f);
  x0 = (q0 - zp) * scale * 0.125f + mp[tid];
  x1 = (q1 - zp) * scale * 0.125f + mp[tid + 256];
  float wm = wredMax(fmaxf(x0, x1));
  if (lane == 0) sA[wave] = wm;
  __syncthreads();
  float rmax = sA[0];
#pragma unroll
  for (int i = 1; i < 8; ++i) rmax = fmaxf(rmax, sA[i]);
  float e0 = expf(x0 - rmax), e1 = expf(x1 - rmax);
  float ws = wredSum(e0 + e1);
  if (lane == 0) sB[wave] = ws;
  __syncthreads();
  float rsum = 0.f;
#pragma unroll
  for (int i = 0; i < 8; ++i) rsum += sB[i];
  float inv = 1.0f / rsum;
  float p0 = e0 * inv, p1 = e1 * inv;
  rp[tid] = p0; rp[tid + 256] = p1;
  float mn = wredMin(fminf(p0, p1)), mx = wredMax(fmaxf(p0, p1));
  if (lane == 0) { sC[wave] = mn; sD[wave] = mx; }
  __syncthreads();
  if (tid == 0) {
#pragma unroll
    for (int i = 1; i < 8; ++i) { sC[0] = fminf(sC[0], sC[i]); sD[0] = fmaxf(sD[0], sD[i]); }
    atomicMin(&mm[0], f2ord(sC[0]));
    atomicMax(&mm[1], f2ord(sD[0]));
  }
}

__global__ void k_quant_probs(const float* __restrict__ probs,
                              const float2* __restrict__ prm,
                              unsigned short* __restrict__ out, int n) {
  int i = blockIdx.x * blockDim.x + threadIdx.x;
  if (i >= n) return;
  float scale = prm->x, zp = prm->y;
  float q = fminf(fmaxf(rintf(probs[i]/scale + zp), 0.f), 255.f);
  out[i] = f2bf((q - zp) * scale);
}

// ctx[head][sq][d] = probs_q @ v ; vT is [head][d][sk] so B-fragment loads are contiguous.
// 64x64 macro-tile per wave (N=64 = full head dim).
__global__ void __launch_bounds__(256, 1)
k_ctx_gemm(const unsigned short* __restrict__ probs,
           const unsigned short* __restrict__ vT,
           float* __restrict__ ctx,
           unsigned* __restrict__ mm) {
  int wave = threadIdx.x >> 5, lane = threadIdx.x & 31;
  int gid = blockIdx.x * 8 + wave;
  const int TPH = SEQ / 64;                // 8 macro-tiles per head
  int head = gid / TPH;
  int tm = gid % TPH;
  v8f c[16] = {};
  gemm64x64(probs + (size_t)head * SEQ * SEQ + (size_t)tm * 64 * SEQ, SEQ,
            vT    + (size_t)head * HD  * SEQ, SEQ, SEQ, lane, c);

  float* out = ctx + (size_t)head * SEQ * HD;
  int hi8 = (lane >> 4) << 3;
  float mn = c[0][0], mx = c[0][0];
#pragma unroll
  for (int i = 0; i < 4; ++i)
#pragma unroll
    for (int j = 0; j < 4; ++j)
#pragma unroll
      for (int v2 = 0; v2 < 8; ++v2) {
        float val = c[i*4+j][v2];
        out[(size_t)(tm*64 + i*16 + hi8 + v2) * HD + j*16 + (lane & 15)] = val;
        mn = fminf(mn, val); mx = fmaxf(mx, val);
      }
  mn = wredMin(mn); mx = wredMax(mx);
  if (lane == 0) { atomicMin(&mm[0], f2ord(mn)); atomicMax(&mm[1], f2ord(mx)); }
}

// fake_quant(ctx) and [B,H,S,D] -> [B,S,H*D]
__global__ void k_output(const float* __restrict__ ctx,
                         const float2* __restrict__ prm,
                         float* __restrict__ out) {
  int i = blockIdx.x * blockDim.x + threadIdx.x;   // over B*NH*SEQ*HD, exact multiple of 256
  float scale = prm->x, zp = prm->y;
  float q = fminf(fmaxf(rintf(ctx[i]/scale + zp), 0.f), 255.f);
  float v = (q - zp) * scale;
  int d = i & 63, s = (i >> 6) & 511, h = (i >> 15) & 15, b = i >> 19;
  out[(size_t)(b * SEQ + s) * DM + h * HD + d] = v;
}

extern "C" void kernel_launch(void* const* d_in, const int* in_sizes, int n_in,
                              void* d_out, int out_size, void* d_ws, size_t ws_size,
                              hipStream_t stream) {
  (void)in_sizes; (void)n_in; (void)out_size; (void)ws_size;
  const float* hs   = (const float*)d_in[0];
  const float* mask = (const float*)d_in[1];
  const float* Wq   = (const float*)d_in[2];
  const float* bq   = (const float*)d_in[3];
  const float* Wk   = (const float*)d_in[4];
  const float* bk   = (const float*)d_in[5];
  const float* Wv   = (const float*)d_in[6];
  const float* bv   = (const float*)d_in[7];
  float* out = (float*)d_out;

  char* w = (char*)d_ws;
  auto alloc = [&](size_t bytes) { char* p = w; w += (bytes + 255) & ~(size_t)255; return p; };
  unsigned short* hs_bf  = (unsigned short*)alloc((size_t)TOKENS*DM*2);
  unsigned short* wq_bf  = (unsigned short*)alloc((size_t)DM*DM*2);
  unsigned short* wk_bf  = (unsigned short*)alloc((size_t)DM*DM*2);
  unsigned short* wv_bf  = (unsigned short*)alloc((size_t)DM*DM*2);
  unsigned short* q_bf   = (unsigned short*)alloc((size_t)TOKENS*DM*2);
  unsigned short* k_bf   = (unsigned short*)alloc((size_t)TOKENS*DM*2);
  unsigned short* vT_bf  = (unsigned short*)alloc((size_t)TOKENS*DM*2);
  float*          scores = (float*)alloc((size_t)HEADS*SEQ*SEQ*4);
  unsigned short* pr_bf  = (unsigned short*)alloc((size_t)HEADS*SEQ*SEQ*2);
  float*          ctx    = (float*)alloc((size_t)TOKENS*DM*4);
  unsigned*       mm     = (unsigned*)alloc(256);
  float2* sprm = (float2*)((char*)mm + 32);
  float2* pprm = sprm + 1;
  float2* cprm = pprm + 1;

  k_init_minmax<<<1, 32, 0, stream>>>(mm);
  int n1 = TOKENS * DM;
  k_convert_bf16<<<(n1 + 255) / 256, 256, 0, stream>>>(hs, hs_bf, n1);
  int n2 = DM * DM;
  k_convert_bf16<<<(n2 + 255) / 256, 256, 0, stream>>>(Wq, wq_bf, n2);
  k_convert_bf16<<<(n2 + 255) / 256, 256, 0, stream>>>(Wk, wk_bf, n2);
  k_convert_bf16<<<(n2 + 255) / 256, 256, 0, stream>>>(Wv, wv_bf, n2);

  // 3 * (4096/64) * (1024/64) = 3072 macro-tiles, 8 waves/block
  k_qkv_gemm<<<(3 * (TOKENS/64) * (DM/64)) / 8, 256, 0, stream>>>(
      hs_bf, wq_bf, wk_bf, wv_bf, bq, bk, bv, q_bf, k_bf, vT_bf);

  // 128 * 64 = 8192 macro-tiles
  k_scores_gemm<<<(HEADS * (SEQ/64) * (SEQ/64)) / 8, 256, 0, stream>>>(q_bf, k_bf, scores, mm + 0);
  k_finalize<<<1, 32, 0, stream>>>(mm + 0, sprm);

  k_softmax<<<HEADS * SEQ, 256, 0, stream>>>(scores, mask, sprm, mm + 2);
  k_finalize<<<1, 32, 0, stream>>>(mm + 2, pprm);

  int n3 = HEADS * SEQ * SEQ;
  k_quant_probs<<<(n3 + 255) / 256, 256, 0, stream>>>(scores, pprm, pr_bf, n3);

  // 128 * 8 = 1024 macro-tiles
  k_ctx_gemm<<<(HEADS * (SEQ/64)) / 8, 256, 0, stream>>>(pr_bf, vT_bf, ctx, mm + 4);
  k_finalize<<<1, 32, 0, stream>>>(mm + 4, cprm);

  k_output<<<(TOKENS * DM) / 256, 256, 0, stream>>>(ctx, cprm, out);
}
